// GCN_17514876633905
// MI455X (gfx1250) — matrix operand order repaired and checked by reading, not compiled, not run on hardware
//
#include <hip/hip_runtime.h>

typedef __attribute__((ext_vector_type(16))) __bf16 v16bf;
typedef __attribute__((ext_vector_type(8)))  float  v8f;

// ---------------------------------------------------------------------------
// small utility kernels
// ---------------------------------------------------------------------------
__global__ void k_fill(float* __restrict__ p, float v, int n) {
    int i = blockIdx.x * blockDim.x + threadIdx.x;
    if (i < n) p[i] = v;
}

__global__ void k_zero4(float4* __restrict__ p, long long n4) {
    long long i = (long long)blockIdx.x * blockDim.x + threadIdx.x;
    if (i < n4) p[i] = make_float4(0.f, 0.f, 0.f, 0.f);
}

__global__ void k_deg(const int* __restrict__ col, float* __restrict__ deg, int E) {
    int i = blockIdx.x * blockDim.x + threadIdx.x;
    if (i < E) atomicAdd(&deg[col[i]], 1.0f);
}

__global__ void k_rsqrt(float* __restrict__ p, int n) {
    int i = blockIdx.x * blockDim.x + threadIdx.x;
    if (i < n) p[i] = rsqrtf(p[i]);   // deg >= 1 always (self loop)
}

// ---------------------------------------------------------------------------
// Pre-convert a weight matrix W[K x Nc] (f32, row-major) into WMMA-B fragment
// order, split into bf16 hi + lo (bf16x3 trick). Fragment tile (kt, nt):
// 32 lanes x 16 bf16, lane n = lane&15 holds K = kt*32 + (lane>>4)*16 + 2j(+1).
// Each lane's 16 bf16 are stored contiguously (32 B) -> b128-pair loads later.
// ---------------------------------------------------------------------------
__global__ void k_prep_wfrag(const float* __restrict__ W, __bf16* __restrict__ hi,
                             __bf16* __restrict__ lo, int K, int Nc) {
    int KT = K / 32, NT = Nc / 16;
    int tid = blockIdx.x * blockDim.x + threadIdx.x;
    if (tid >= KT * NT * 32) return;
    int lane = tid & 31;
    int t    = tid >> 5;
    int nt   = t % NT;
    int kt   = t / NT;
    int n    = nt * 16 + (lane & 15);
    int khi  = lane >> 4;
    size_t base = ((size_t)t * 32 + lane) * 16;
    for (int j = 0; j < 8; ++j) {
        int k = kt * 32 + khi * 16 + 2 * j;
        for (int s = 0; s < 2; ++s) {
            float w   = W[(size_t)(k + s) * Nc + n];
            __bf16 wh = (__bf16)w;
            hi[base + 2 * j + s] = wh;
            lo[base + 2 * j + s] = (__bf16)(w - (float)wh);
        }
    }
}

// ---------------------------------------------------------------------------
// GEMM: C[M x Nc] = A[M x K] * W   (W given as bf16 hi/lo fragments)
// One wave per 16-row tile; NT 16x16 f32 accumulators held in VGPRs.
// bf16x3: acc += Ahi*Bhi + Ahi*Blo + Alo*Bhi  (fp32-comparable accuracy).
// ---------------------------------------------------------------------------
template <int KT, int NT>
__global__ __launch_bounds__(256)
void k_gemm(const float* __restrict__ A, const __bf16* __restrict__ Bhi,
            const __bf16* __restrict__ Blo, float* __restrict__ C, int rowTiles) {
    constexpr int K  = KT * 32;
    constexpr int Nc = NT * 16;
    int wave = threadIdx.x >> 5;
    int lane = threadIdx.x & 31;
    int tile = blockIdx.x * 8 + wave;
    if (tile >= rowTiles) return;          // uniform per wave -> EXEC all 1s below

    int rowBase = tile * 16;
    int m   = lane & 15;
    int khi = lane >> 4;
    const float* rowA = A + (size_t)(rowBase + m) * K;

    v8f acc[NT];
    for (int nt = 0; nt < NT; ++nt)
        for (int r = 0; r < 8; ++r) acc[nt][r] = 0.f;

    for (int kt = 0; kt < KT; ++kt) {
        // A fragment: lane holds K = kt*32 + khi*16 + {0..15}
        const float* p = rowA + kt * 32 + khi * 16;
        float av[16];
        float4 f0 = *(const float4*)(p);
        float4 f1 = *(const float4*)(p + 4);
        float4 f2 = *(const float4*)(p + 8);
        float4 f3 = *(const float4*)(p + 12);
        av[0] = f0.x;  av[1] = f0.y;  av[2]  = f0.z;  av[3]  = f0.w;
        av[4] = f1.x;  av[5] = f1.y;  av[6]  = f1.z;  av[7]  = f1.w;
        av[8] = f2.x;  av[9] = f2.y;  av[10] = f2.z;  av[11] = f2.w;
        av[12] = f3.x; av[13] = f3.y; av[14] = f3.z;  av[15] = f3.w;
        v16bf ahi, alo;
        for (int j = 0; j < 16; ++j) {
            __bf16 h = (__bf16)av[j];
            ahi[j] = h;
            alo[j] = (__bf16)(av[j] - (float)h);
        }
        for (int nt = 0; nt < NT; ++nt) {
            size_t fb = ((size_t)(kt * NT + nt) * 32 + lane) * 16;
            v16bf bhi = *(const v16bf*)(Bhi + fb);
            v16bf blo = *(const v16bf*)(Blo + fb);
            acc[nt] = __builtin_amdgcn_wmma_f32_16x16x32_bf16(
                false, ahi, false, bhi, (short)0, acc[nt], false, false);
            acc[nt] = __builtin_amdgcn_wmma_f32_16x16x32_bf16(
                false, ahi, false, blo, (short)0, acc[nt], false, false);
            acc[nt] = __builtin_amdgcn_wmma_f32_16x16x32_bf16(
                false, alo, false, bhi, (short)0, acc[nt], false, false);
        }
    }
    // C/D layout: VGPR r, lanes 0-15 -> M=r, lanes 16-31 -> M=8+r; N = lane&15
    for (int nt = 0; nt < NT; ++nt) {
        int n = nt * 16 + m;
        for (int r = 0; r < 8; ++r) {
            int mr = rowBase + r + khi * 8;
            C[(size_t)mr * Nc + n] = acc[nt][r];
        }
    }
}

// ---------------------------------------------------------------------------
// Edge aggregation: agg[col[e]] += h[row[e]] * dis[row]*dis[col]
// D/4 lanes per edge, float4 gather + 4 f32 atomics (L2-resident accumulator).
// ---------------------------------------------------------------------------
template <int D>
__global__ void k_aggregate(const int* __restrict__ row, const int* __restrict__ col,
                            const float* __restrict__ dis, const float* __restrict__ h,
                            float* __restrict__ agg, int E) {
    constexpr int LPE = D / 4;
    long long tid = (long long)blockIdx.x * blockDim.x + threadIdx.x;
    int e = (int)(tid / LPE);
    int l = (int)(tid % LPE);
    if (e >= E) return;
    int r = row[e], c = col[e];
    float nrm = dis[r] * dis[c];
    float4 v = *(const float4*)(h + (size_t)r * D + l * 4);
    float* dst = agg + (size_t)c * D + l * 4;
    atomicAdd(dst + 0, v.x * nrm);
    atomicAdd(dst + 1, v.y * nrm);
    atomicAdd(dst + 2, v.z * nrm);
    atomicAdd(dst + 3, v.w * nrm);
}

// layer-1 finalize: a1 = relu(agg + h*dis^2 (self loop) + b), in place in agg
template <int D>
__global__ void k_final_relu(float* __restrict__ agg, const float* __restrict__ h,
                             const float* __restrict__ dis, const float* __restrict__ b,
                             int Nn) {
    constexpr int LPR = D / 4;
    long long tid = (long long)blockIdx.x * blockDim.x + threadIdx.x;
    if (tid >= (long long)Nn * LPR) return;
    int i = (int)(tid / LPR);
    int l = (int)(tid % LPR);
    float dd = dis[i];
    float sl = dd * dd;
    float4 a  = *(float4*)(agg + (size_t)i * D + l * 4);
    float4 hv = *(const float4*)(h + (size_t)i * D + l * 4);
    float4 bv = *(const float4*)(b + l * 4);
    float4 o;
    o.x = fmaxf(a.x + hv.x * sl + bv.x, 0.f);
    o.y = fmaxf(a.y + hv.y * sl + bv.y, 0.f);
    o.z = fmaxf(a.z + hv.z * sl + bv.z, 0.f);
    o.w = fmaxf(a.w + hv.w * sl + bv.w, 0.f);
    *(float4*)(agg + (size_t)i * D + l * 4) = o;
}

// layer-2 finalize + log_softmax over 64 cols; one wave32 per row, 2 elems/lane
__global__ void k_final_logsoftmax(const float* __restrict__ agg, const float* __restrict__ h,
                                   const float* __restrict__ dis, const float* __restrict__ b,
                                   float* __restrict__ out, int Nn) {
    int wid  = threadIdx.x >> 5;
    int lane = threadIdx.x & 31;
    int i = blockIdx.x * 8 + wid;
    if (i >= Nn) return;
    float dd = dis[i];
    float sl = dd * dd;
    const float* ap = agg + (size_t)i * 64;
    const float* hp = h + (size_t)i * 64;
    float v0 = ap[lane]      + hp[lane]      * sl + b[lane];
    float v1 = ap[lane + 32] + hp[lane + 32] * sl + b[lane + 32];
    float mx = fmaxf(v0, v1);
    for (int off = 16; off; off >>= 1) mx = fmaxf(mx, __shfl_xor(mx, off, 32));
    float s = expf(v0 - mx) + expf(v1 - mx);
    for (int off = 16; off; off >>= 1) s += __shfl_xor(s, off, 32);
    float lse = logf(s) + mx;
    out[(size_t)i * 64 + lane]      = v0 - lse;
    out[(size_t)i * 64 + lane + 32] = v1 - lse;
}

// ---------------------------------------------------------------------------
extern "C" void kernel_launch(void* const* d_in, const int* in_sizes, int n_in,
                              void* d_out, int out_size, void* d_ws, size_t ws_size,
                              hipStream_t stream) {
    const float* x  = (const float*)d_in[0];
    const int*   ei = (const int*)d_in[1];
    const float* W1 = (const float*)d_in[2];
    const float* b1 = (const float*)d_in[3];
    const float* W2 = (const float*)d_in[4];
    const float* b2 = (const float*)d_in[5];

    const int N = in_sizes[0] / 256;   // 100000
    const int E = in_sizes[1] / 2;     // 1600000
    const int* row = ei;               // source
    const int* col = ei + E;           // destination

    // workspace carve-up (floats)
    float* ws = (float*)d_ws;
    size_t o = 0;
    float* dis  = ws + o; o += ((size_t)N + 1023) & ~(size_t)1023;
    float* h1   = ws + o; o += (size_t)N * 128;
    float* agg1 = ws + o; o += (size_t)N * 128;   // becomes a1 in place
    float* h2   = ws + o; o += (size_t)N * 64;
    float* agg2 = ws + o; o += (size_t)N * 64;
    __bf16* w1hi = (__bf16*)(ws + o);
    __bf16* w1lo = w1hi + 8 * 8 * 512;            // (256/32)*(128/16)*32*16
    __bf16* w2hi = w1lo + 8 * 8 * 512;
    __bf16* w2lo = w2hi + 4 * 4 * 512;            // (128/32)*(64/16)*32*16

    // --- norm: deg (with self loop) -> dis = rsqrt(deg)
    k_fill<<<(N + 255) / 256, 256, 0, stream>>>(dis, 1.0f, N);
    k_deg<<<(E + 255) / 256, 256, 0, stream>>>(col, dis, E);
    k_rsqrt<<<(N + 255) / 256, 256, 0, stream>>>(dis, N);

    // --- weight fragments (bf16 hi/lo, WMMA-B layout)
    k_prep_wfrag<<<(8 * 8 * 32 + 255) / 256, 256, 0, stream>>>(W1, w1hi, w1lo, 256, 128);
    k_prep_wfrag<<<(4 * 4 * 32 + 255) / 256, 256, 0, stream>>>(W2, w2hi, w2lo, 128, 64);

    // --- layer 1
    int rowTiles = N / 16;  // 6250
    k_gemm<8, 8><<<(rowTiles + 7) / 8, 256, 0, stream>>>(x, w1hi, w1lo, h1, rowTiles);
    long long n4a = (long long)N * 128 / 4;
    k_zero4<<<(unsigned)((n4a + 255) / 256), 256, 0, stream>>>((float4*)agg1, n4a);
    long long t1 = (long long)E * 32;
    k_aggregate<128><<<(unsigned)((t1 + 255) / 256), 256, 0, stream>>>(row, col, dis, h1, agg1, E);
    long long f1 = (long long)N * 32;
    k_final_relu<128><<<(unsigned)((f1 + 255) / 256), 256, 0, stream>>>(agg1, h1, dis, b1, N);

    // --- layer 2
    k_gemm<4, 4><<<(rowTiles + 7) / 8, 256, 0, stream>>>(agg1, w2hi, w2lo, h2, rowTiles);
    long long n4b = (long long)N * 64 / 4;
    k_zero4<<<(unsigned)((n4b + 255) / 256), 256, 0, stream>>>((float4*)agg2, n4b);
    long long t2 = (long long)E * 16;
    k_aggregate<64><<<(unsigned)((t2 + 255) / 256), 256, 0, stream>>>(row, col, dis, h2, agg2, E);
    k_final_logsoftmax<<<(N + 7) / 8, 256, 0, stream>>>(agg2, h2, dis, b2, (float*)d_out, N);
}
